// GCNEncoder_8873402434235
// MI455X (gfx1250) — matrix operand order repaired and checked by reading, not compiled, not run on hardware
//
#include <hip/hip_runtime.h>

// GCN encoder for MI455X (gfx1250, wave32).
// GEMMs via v_wmma_f32_16x16x32_bf16 (B-fragments preloaded in registers);
// edge scatter via global_atomic_add_f32.

#define BN_EPS 1e-5f

typedef __attribute__((ext_vector_type(16))) __bf16 v16bf;
typedef __attribute__((ext_vector_type(8)))  float  v8f;

struct __attribute__((aligned(16))) F4 { float x, y, z, w; };
struct __attribute__((aligned(8)))  F2 { float x, y; };
struct __attribute__((aligned(16))) U4 { unsigned int x, y, z, w; };

union Frag16 {
  v16bf v;
  unsigned short u[16];
  U4 q[2];
};

__device__ __forceinline__ unsigned short f2bf(float f) {
  unsigned int u = __float_as_uint(f);
  unsigned int r = u + 0x7FFFu + ((u >> 16) & 1u);   // round-to-nearest-even
  return (unsigned short)(r >> 16);
}

__device__ __forceinline__ void atomAddF(float* p, float v) {
  __hip_atomic_fetch_add(p, v, __ATOMIC_RELAXED, __HIP_MEMORY_SCOPE_AGENT);
}

// ---------------- degree / dinv ----------------
__global__ void k_deg(const long long* __restrict__ dst, float* __restrict__ deg,
                      long long E) {
  long long e = (long long)blockIdx.x * blockDim.x + threadIdx.x;
  if (e < E) atomAddF(&deg[(int)dst[e]], 1.0f);
}

__global__ void k_dinv(float* __restrict__ deg, int n) {
  int i = blockIdx.x * blockDim.x + threadIdx.x;
  if (i < n) deg[i] = rsqrtf(deg[i] + 1.0f);   // +1 self loop; always > 0
}

// ---------------- column stats of x [n,128] ----------------
__global__ void k_colstats128(const float* __restrict__ x, float* __restrict__ sums,
                              float* __restrict__ sumsq, int n) {
  int c = threadIdx.x;              // 0..127
  int r0 = blockIdx.x * 128;
  float s = 0.f, s2 = 0.f;
  for (int r = 0; r < 128; ++r) {
    int row = r0 + r;
    if (row < n) {
      float v = x[(size_t)row * 128 + c];
      s += v; s2 += v * v;
    }
  }
  atomAddF(&sums[c], s);
  atomAddF(&sumsq[c], s2);
}

// ---------------- BN finalize: a = g*rsqrt(var+eps), c = b - mu*a ----------------
__global__ void k_finalize(const float* __restrict__ sums, const float* __restrict__ sumsq,
                           const float* __restrict__ g, const float* __restrict__ b,
                           float* __restrict__ a, float* __restrict__ c,
                           int dim, float invn) {
  int i = threadIdx.x;
  if (i < dim) {
    float mu  = sums[i] * invn;
    float var = sumsq[i] * invn - mu * mu;
    float av  = g[i] * rsqrtf(var + BN_EPS);
    a[i] = av;
    c[i] = b[i] - mu * av;
  }
}

// ---------------- folded bias for GEMM1: bpf[n] = bp[n] + sum_k c_in[k]*Wp[k,n] ----------------
__global__ void k_bias_fold(const float* __restrict__ Wp, const float* __restrict__ cin,
                            const float* __restrict__ bp, float* __restrict__ bout) {
  int n = threadIdx.x;  // 64
  float s = bp[n];
  for (int k = 0; k < 128; ++k) s += cin[k] * Wp[k * 64 + n];
  bout[n] = s;
}

// ---------------- pack W[K,64] (optionally row-scaled) into WMMA B-fragments ----------------
// B layout per ISA (16-bit B, 32x16): lane l -> col n = cg*16 + (l&15); slot i -> K = kc*32 + 8*(l>>4) + (i&7) + 16*(i>>3)
__global__ void k_pack(const float* __restrict__ W, const float* __restrict__ scale,
                       unsigned short* __restrict__ Bp, int kchunks) {
  int idx = blockIdx.x * blockDim.x + threadIdx.x;
  int total = kchunks * 4 * 32 * 16;
  if (idx >= total) return;
  int i   = idx & 15;
  int l   = (idx >> 4) & 31;
  int cg  = (idx >> 9) & 3;
  int kc  = idx >> 11;
  int n   = cg * 16 + (l & 15);
  int k   = kc * 32 + (l >> 4) * 8 + (i & 7) + (i >> 3) * 16;
  float v = W[k * 64 + n];
  if (scale) v *= scale[k];
  Bp[idx] = f2bf(v);
}

// ---------------- WMMA GEMM: out[nrows,64] = A[nrows,K] @ Wpack (+bias)(+relu) ----------------
// Block = 256 threads = 8 waves; wave -> 16 rows x 64 cols; K fully unrolled.
// All B-fragments preloaded into registers before the K-loop so the only
// loop-carried memory dependency is the A-tile load.
template <int KCHUNKS>
__global__ __launch_bounds__(256)
void k_gemm(const float* __restrict__ A, const unsigned short* __restrict__ Bpack,
            const float* __restrict__ bias, float* __restrict__ out,
            int nrows, int K, int do_relu) {
  const int tid  = threadIdx.x;
  const int wave = tid >> 5;
  const int lane = tid & 31;
  const int half = lane >> 4;
  const int m    = lane & 15;
  const int rowbase = blockIdx.x * 128 + wave * 16;
  const int arow = rowbase + m;
  const bool rvalid = arow < nrows;
  const float* ap = A + (size_t)arow * K;

  // Preload all weight fragments (KCHUNKS*4 fragments, 8 VGPRs each).
  Frag16 bfr[KCHUNKS][4];
  #pragma unroll
  for (int kc = 0; kc < KCHUNKS; ++kc) {
    #pragma unroll
    for (int cg = 0; cg < 4; ++cg) {
      const U4* bq = reinterpret_cast<const U4*>(Bpack) +
                     ((size_t)(kc * 4 + cg) * 32 + lane) * 2;
      bfr[kc][cg].q[0] = bq[0];
      bfr[kc][cg].q[1] = bq[1];
    }
  }

  v8f acc[4] = {};

  #pragma unroll
  for (int kc = 0; kc < KCHUNKS; ++kc) {
    // A fragment (16-bit A 16x32 layout): slots 0..7 -> K = kb..kb+7, slots 8..15 -> K = kb+16..kb+23
    Frag16 af;
    const int kb = kc * 32 + 8 * half;
    if (rvalid) {
      F4 q0 = *(const F4*)(ap + kb);
      F4 q1 = *(const F4*)(ap + kb + 4);
      F4 q2 = *(const F4*)(ap + kb + 16);
      F4 q3 = *(const F4*)(ap + kb + 20);
      af.u[0]  = f2bf(q0.x); af.u[1]  = f2bf(q0.y); af.u[2]  = f2bf(q0.z); af.u[3]  = f2bf(q0.w);
      af.u[4]  = f2bf(q1.x); af.u[5]  = f2bf(q1.y); af.u[6]  = f2bf(q1.z); af.u[7]  = f2bf(q1.w);
      af.u[8]  = f2bf(q2.x); af.u[9]  = f2bf(q2.y); af.u[10] = f2bf(q2.z); af.u[11] = f2bf(q2.w);
      af.u[12] = f2bf(q3.x); af.u[13] = f2bf(q3.y); af.u[14] = f2bf(q3.z); af.u[15] = f2bf(q3.w);
    } else {
      #pragma unroll
      for (int i = 0; i < 16; ++i) af.u[i] = 0;
    }

    #pragma unroll
    for (int cg = 0; cg < 4; ++cg) {
      acc[cg] = __builtin_amdgcn_wmma_f32_16x16x32_bf16(
          false, af.v, false, bfr[kc][cg].v, (short)0, acc[cg], false, false);
    }
  }

  // D layout: lane -> col n = cg*16 + m; vgpr r -> row = rowbase + 8*half + r
  #pragma unroll
  for (int cg = 0; cg < 4; ++cg) {
    #pragma unroll
    for (int r = 0; r < 8; ++r) {
      int ro = rowbase + half * 8 + r;
      if (ro < nrows) {
        int col = cg * 16 + m;
        float v = acc[cg][r];
        if (bias) v += bias[col];
        if (do_relu) v = fmaxf(v, 0.f);
        out[(size_t)ro * 64 + col] = v;
      }
    }
  }
}

// ---------------- edge scatter: out[dst] += dinv[s]*dinv[d] * g[src]  (one wave per edge) ----------------
__global__ __launch_bounds__(256)
void k_scatter(const long long* __restrict__ src, const long long* __restrict__ dst,
               const float* __restrict__ dinv, const float* __restrict__ g,
               float* __restrict__ out, long long E) {
  long long gid = (long long)blockIdx.x * blockDim.x + threadIdx.x;
  long long e = gid >> 5;
  int lane = (int)(gid & 31);
  if (e < E) {
    int s = (int)src[e];
    int d = (int)dst[e];
    float norm = dinv[s] * dinv[d];
    F2 v = *(const F2*)(g + (size_t)s * 64 + lane * 2);
    float* o = out + (size_t)d * 64 + lane * 2;
    atomAddF(o,     v.x * norm);
    atomAddF(o + 1, v.y * norm);
  }
}

// ---------------- self-loop + bias + BN-stat accumulation (64 rows per block, dim=64) ----------------
__global__ __launch_bounds__(256)
void k_selfloop(float* __restrict__ acc, const float* __restrict__ g,
                const float* __restrict__ dinv, const float* __restrict__ bias,
                float* __restrict__ sums, float* __restrict__ sumsq, int n) {
  __shared__ float shs[256];
  __shared__ float shs2[256];
  int t = threadIdx.x;
  int c = t & 63;
  int q = t >> 6;           // 0..3
  int i0 = blockIdx.x * 64;
  float bc = bias[c];
  float s = 0.f, s2 = 0.f;
  for (int k = 0; k < 16; ++k) {
    int i = i0 + q * 16 + k;
    if (i < n) {
      float dv = dinv[i];
      size_t off = (size_t)i * 64 + c;
      float v = acc[off] + dv * dv * g[off] + bc;
      acc[off] = v;
      s += v; s2 += v * v;
    }
  }
  shs[t] = s; shs2[t] = s2;
  __syncthreads();
  if (t < 64) {
    s  = shs[t]  + shs[t + 64]  + shs[t + 128]  + shs[t + 192];
    s2 = shs2[t] + shs2[t + 64] + shs2[t + 128] + shs2[t + 192];
    atomAddF(&sums[t], s);
    atomAddF(&sumsq[t], s2);
  }
}

// ---------------- elementwise BN apply (in place), optional relu ----------------
__global__ void k_bn_apply(float* __restrict__ buf, const float* __restrict__ a,
                           const float* __restrict__ c, size_t total, int relu) {
  size_t i = (size_t)blockIdx.x * blockDim.x + threadIdx.x;
  if (i < total) {
    int col = (int)(i & 63);
    float v = a[col] * buf[i] + c[col];
    if (relu) v = fmaxf(v, 0.f);
    buf[i] = v;
  }
}

extern "C" void kernel_launch(void* const* d_in, const int* in_sizes, int n_in,
                              void* d_out, int out_size, void* d_ws, size_t ws_size,
                              hipStream_t stream) {
  const float*     x    = (const float*)d_in[0];
  const long long* ei   = (const long long*)d_in[1];
  const float*     bng  = (const float*)d_in[2];
  const float*     bnb  = (const float*)d_in[3];
  const float*     Wp   = (const float*)d_in[4];
  const float*     bp   = (const float*)d_in[5];
  const float*     W1   = (const float*)d_in[6];
  const float*     b1   = (const float*)d_in[7];
  const float*     bn1g = (const float*)d_in[8];
  const float*     bn1b = (const float*)d_in[9];
  const float*     W2   = (const float*)d_in[10];
  const float*     b2   = (const float*)d_in[11];
  const float*     bn2g = (const float*)d_in[12];
  const float*     bn2b = (const float*)d_in[13];

  const int N = in_sizes[0] / 128;
  const long long E = (long long)in_sizes[1] / 2;
  const long long* srcp = ei;
  const long long* dstp = ei + E;
  float* out = (float*)d_out;

  // ---- workspace carve (all offsets multiples of 4 floats = 16B) ----
  float* ws = (float*)d_ws;
  size_t off = 0;
  float* deg  = ws + off; off += (size_t)((N + 3) & ~3);   // becomes dinv in place
  float* stat = ws + off; off += 512;  // sums_in|sumsq_in|sums1|sumsq1|sums2|sumsq2
  float* a_in = ws + off; off += 128;
  float* c_in = ws + off; off += 128;
  float* a1   = ws + off; off += 64;
  float* c1   = ws + off; off += 64;
  float* a2   = ws + off; off += 64;
  float* c2   = ws + off; off += 64;
  float* bpf  = ws + off; off += 64;
  unsigned short* Bp1 = (unsigned short*)(ws + off); off += 4096;  // 8192 shorts (K=128)
  unsigned short* Bp2 = (unsigned short*)(ws + off); off += 2048;  // 4096 shorts (K=64)
  unsigned short* Bp3 = (unsigned short*)(ws + off); off += 2048;
  float* B0 = ws + off; off += (size_t)N * 64;
  float* B1 = ws + off; off += (size_t)N * 64;

  float* sums_in = stat;       float* sumsq_in = stat + 128;
  float* sums1   = stat + 256; float* sumsq1   = stat + 320;
  float* sums2   = stat + 384; float* sumsq2   = stat + 448;

  const float invn = 1.0f / (float)N;
  const int tb = 256;
  const int gemmBlocks = (N + 127) / 128;
  const size_t total = (size_t)N * 64;
  const int sblocks = (int)((32LL * E + 255) / 256);

  // zero accumulators (deterministic per call)
  hipMemsetAsync(deg, 0, (size_t)N * 4, stream);
  hipMemsetAsync(stat, 0, 512 * 4, stream);

  // degrees + dinv
  k_deg<<<(int)((E + tb - 1) / tb), tb, 0, stream>>>(dstp, deg, E);
  k_dinv<<<(N + tb - 1) / tb, tb, 0, stream>>>(deg, N);

  // input BN stats -> affine; fold into Wp/bias; pack all weights to WMMA fragments
  k_colstats128<<<(N + 127) / 128, 128, 0, stream>>>(x, sums_in, sumsq_in, N);
  k_finalize<<<1, 128, 0, stream>>>(sums_in, sumsq_in, bng, bnb, a_in, c_in, 128, invn);
  k_bias_fold<<<1, 64, 0, stream>>>(Wp, c_in, bp, bpf);
  k_pack<<<(8192 + tb - 1) / tb, tb, 0, stream>>>(Wp, a_in, Bp1, 4);
  k_pack<<<(4096 + tb - 1) / tb, tb, 0, stream>>>(W1, nullptr, Bp2, 2);
  k_pack<<<(4096 + tb - 1) / tb, tb, 0, stream>>>(W2, nullptr, Bp3, 2);

  // h0 = relu(BN(x)@Wp + bp)   [WMMA, K=128]
  k_gemm<4><<<gemmBlocks, 256, 0, stream>>>(x, Bp1, bpf, B0, N, 128, 1);
  // g1 = h0 @ W1               [WMMA, K=64]
  k_gemm<2><<<gemmBlocks, 256, 0, stream>>>(B0, Bp2, nullptr, B1, N, 64, 0);

  // conv1: scatter + self-loop + bias, fused BN stats
  hipMemsetAsync(B0, 0, total * 4, stream);
  k_scatter<<<sblocks, 256, 0, stream>>>(srcp, dstp, deg, B1, B0, E);
  k_selfloop<<<(N + 63) / 64, 256, 0, stream>>>(B0, B1, deg, b1, sums1, sumsq1, N);
  k_finalize<<<1, 64, 0, stream>>>(sums1, sumsq1, bn1g, bn1b, a1, c1, 64, invn);
  k_bn_apply<<<(int)((total + 255) / 256), 256, 0, stream>>>(B0, a1, c1, total, 1);

  // g2 = h1 @ W2               [WMMA, K=64]
  k_gemm<2><<<gemmBlocks, 256, 0, stream>>>(B0, Bp3, nullptr, B1, N, 64, 0);

  // conv2: scatter into d_out + self-loop + bias, fused BN stats, final BN
  hipMemsetAsync(out, 0, total * 4, stream);
  k_scatter<<<sblocks, 256, 0, stream>>>(srcp, dstp, deg, B1, out, E);
  k_selfloop<<<(N + 63) / 64, 256, 0, stream>>>(out, B1, deg, b2, sums2, sumsq2, N);
  k_finalize<<<1, 64, 0, stream>>>(sums2, sumsq2, bn2g, bn2b, a2, c2, 64, invn);
  k_bn_apply<<<(int)((total + 255) / 256), 256, 0, stream>>>(out, a2, c2, total, 0);

  (void)n_in; (void)out_size; (void)ws_size;
}
